// MSDeformAttn_20882130993648
// MI455X (gfx1250) — matrix operand order repaired and checked by reading, not compiled, not run on hardware
//
#include <hip/hip_runtime.h>

#define NN 2
#define TT 3
#define CIN 128
#define HH 48
#define WW 48
#define HW 2304          // HH*WW
#define GG 6             // NN*TT
#define NHEAD 8
#define NLVL 3
#define NPTS 4
#define DHEAD 16
#define LQTOT 6912       // TT*HW
#define KTOT 1152        // CIN*9

typedef __attribute__((ext_vector_type(16))) __bf16 v16bf;
typedef __attribute__((ext_vector_type(8)))  float  v8f;

// -------- implicit-GEMM 3x3 SAME conv via WMMA bf16 ----------------------
// X: (GG, CIN, HH, WW) f32, Wt: (Cout, CIN, 3, 3) f32, Y: (GG, Cout, HH, WW)
// K ordering: k = s*128 + ci, s = kh*3+kw  (one tap per 128-wide K band).
__global__ __launch_bounds__(256)
void conv3x3_wmma(const float* __restrict__ X, const float* __restrict__ Wt,
                  const float* __restrict__ Bias, float* __restrict__ Y,
                  int Cout) {
  __shared__ __bf16 wlds[16 * KTOT];           // [n][k] bf16, 36 KB

  const int tid  = threadIdx.x;
  const int lane = tid & 31;
  const int wave = tid >> 5;
  const int hi   = (lane >> 4) & 1;            // half-wave select
  const int n16  = lane & 15;
  const int coBase = blockIdx.y * 16;

  // stage this block's 16 output-channel weight slab into LDS as bf16
  for (int i = tid; i < 16 * KTOT; i += 256) {
    int nn = i / KTOT;
    int k  = i - nn * KTOT;
    int s  = k >> 7;                 // 0..8 tap
    int ci = k & 127;
    wlds[i] = (__bf16)Wt[((coBase + nn) * CIN + ci) * 9 + s];
  }
  __syncthreads();

  const int mTile = blockIdx.x * 8 + wave;     // 16-pixel tile per wave
  const int m  = mTile * 16 + n16;             // A-row this lane feeds
  const int g  = m / HW;
  const int hw = m - g * HW;
  const int h  = hw / WW;
  const int w  = hw - h * WW;
  const float* xg = X + (size_t)g * CIN * HW;

  union { v8f v; float f[8]; } c = {};

  const unsigned* ldsw = (const unsigned*)wlds;
  const int bRow = n16 * (KTOT >> 1);          // u32 row base for this N column

  for (int s = 0; s < 9; ++s) {
    const int kh = s / 3, kw = s - kh * 3;
    const int ih = h + kh - 1, iw = w + kw - 1;
    const bool inb = (ih >= 0) && (ih < HH) && (iw >= 0) && (iw < WW);
    const int   pixoff = inb ? (ih * WW + iw) : 0;   // clamp to safe address
    const float fm     = inb ? 1.0f : 0.0f;          // zero OOB via multiply
    const float* xp = xg + pixoff;

#pragma unroll
    for (int kc = 0; kc < 4; ++kc) {
      const int cbase = kc * 32 + hi * 8;            // channel base, this half

      // A fragment: lane = M row; unconditional, mask-multiplied loads
      union { v16bf v; __bf16 b[16]; } a;
#pragma unroll
      for (int e = 0; e < 16; ++e) {
        int kb = (e < 8) ? e : e + 8;                // K within half: 0-7,16-23
        a.b[e] = (__bf16)(xp[(cbase + kb) * HW] * fm);
      }

      // B fragment from LDS: lane = N column, half-waves hold K 0-7,16-23 / 8-15,24-31
      union { v16bf v; unsigned u32[8]; } b;
      const int kbb = s * 128 + kc * 32 + hi * 8;
#pragma unroll
      for (int e2 = 0; e2 < 8; ++e2)
        b.u32[e2] = ldsw[bRow + ((kbb + 2 * e2 + (e2 >= 4 ? 8 : 0)) >> 1)];

      c.v = __builtin_amdgcn_wmma_f32_16x16x32_bf16(
          false, a.v, false, b.v, (short)0, c.v, false, false);
    }
  }

  const float bias = Bias[coBase + n16];
#pragma unroll
  for (int r = 0; r < 8; ++r) {
    int mm = mTile * 16 + r + hi * 8;          // D layout: M = r + 8*(lane>=16)
    int g2  = mm / HW;
    int hw2 = mm - g2 * HW;
    Y[((size_t)g2 * Cout + coBase + n16) * HW + hw2] = c.f[r] + bias;
  }
}

// -------- flow composition: ff02 = ff01 + mask*warp(ff12, ff01), etc. ----
__device__ __forceinline__ float bilin_clamp(const float* img, float px, float py) {
  float x0f = floorf(px), y0f = floorf(py);
  int x0 = (int)x0f, y0 = (int)y0f;
  float wx = px - x0f, wy = py - y0f;
  int x0c = min(max(x0, 0), WW - 1), x1c = min(max(x0 + 1, 0), WW - 1);
  int y0c = min(max(y0, 0), HH - 1), y1c = min(max(y0 + 1, 0), HH - 1);
  float v00 = img[y0c * WW + x0c], v10 = img[y0c * WW + x1c];
  float v01 = img[y1c * WW + x0c], v11 = img[y1c * WW + x1c];
  return v00*(1-wx)*(1-wy) + v10*wx*(1-wy) + v01*(1-wx)*wy + v11*wx*wy;
}

__device__ __forceinline__ float bilin_mask(float px, float py) {
  float x0f = floorf(px), y0f = floorf(py);
  int x0 = (int)x0f, y0 = (int)y0f;
  float wx = px - x0f, wy = py - y0f;
  auto ib = [](int x, int y) {
    return (x >= 0 && x < WW && y >= 0 && y < HH) ? 1.0f : 0.0f;
  };
  return ib(x0,y0)*(1-wx)*(1-wy) + ib(x0+1,y0)*wx*(1-wy)
       + ib(x0,y0+1)*(1-wx)*wy   + ib(x0+1,y0+1)*wx*wy;
}

__global__ void flow_compose(const float* __restrict__ FF, const float* __restrict__ FB,
                             float* __restrict__ ff02, float* __restrict__ fb20) {
  int idx = blockIdx.x * blockDim.x + threadIdx.x;
  if (idx >= NN * HW) return;
  int n = idx / HW, hw = idx - n * HW;
  int h = hw / WW, w = hw - h * WW;
  const float* ff01 = FF + (size_t)n * 4 * HW;   // (2ch, HW)
  const float* ff12 = ff01 + 2 * HW;
  const float* fb10 = FB + (size_t)n * 4 * HW;
  const float* fb21 = fb10 + 2 * HW;
  {
    float vx = (float)w + ff01[hw];
    float vy = (float)h + ff01[HW + hw];
    float mm = (bilin_mask(vx, vy) < 0.999f) ? 0.f : 1.f;
    ff02[(size_t)n*2*HW + hw]      = ff01[hw]      + mm * bilin_clamp(ff12,      vx, vy);
    ff02[(size_t)n*2*HW + HW + hw] = ff01[HW + hw] + mm * bilin_clamp(ff12 + HW, vx, vy);
  }
  {
    float vx = (float)w + fb21[hw];
    float vy = (float)h + fb21[HW + hw];
    float mm = (bilin_mask(vx, vy) < 0.999f) ? 0.f : 1.f;
    fb20[(size_t)n*2*HW + hw]      = fb21[hw]      + mm * bilin_clamp(fb10,      vx, vy);
    fb20[(size_t)n*2*HW + HW + hw] = fb21[HW + hw] + mm * bilin_clamp(fb10 + HW, vx, vy);
  }
}

// -------- softmax + flow-guided deformable sampling ----------------------
// one thread per (n, q, head); writes (G, C, H, W) so the final conv can
// consume it directly.
__global__ __launch_bounds__(256)
void deform_sample(const float* __restrict__ Vconv, const float* __restrict__ Oconv,
                   const float* __restrict__ Aconv, const float* __restrict__ Ref,
                   const float* __restrict__ FF, const float* __restrict__ FB,
                   const float* __restrict__ ff02, const float* __restrict__ fb20,
                   float* __restrict__ Out) {
  int idx = blockIdx.x * blockDim.x + threadIdx.x;
  if (idx >= NN * LQTOT * NHEAD) return;
  int head = idx & (NHEAD - 1);
  int q = (idx >> 3) % LQTOT;
  int n = idx / (LQTOT * NHEAD);
  int t = q / HW, hw = q - t * HW;
  int h = hw / WW, w = hw - h * WW;
  int g = n * TT + t;

  // softmax over NL*NPT = 12 logits for this head
  float lg[12];
  const float* ap = Aconv + ((size_t)g * (NHEAD*NLVL*NPTS) + head * (NLVL*NPTS)) * HW + hw;
#pragma unroll
  for (int j = 0; j < 12; ++j) lg[j] = ap[j * HW];
  float mx = lg[0];
#pragma unroll
  for (int j = 1; j < 12; ++j) mx = fmaxf(mx, lg[j]);
  float ssum = 0.f;
#pragma unroll
  for (int j = 0; j < 12; ++j) { lg[j] = __expf(lg[j] - mx); ssum += lg[j]; }
  float inv = 1.f / ssum;

  float acc[DHEAD];
#pragma unroll
  for (int d = 0; d < DHEAD; ++d) acc[d] = 0.f;

  for (int l = 0; l < NLVL; ++l) {
    // flow-guided additive offset, chosen by (frame t, level l)
    const float* addp = nullptr;
    if (t == 0)      { if (l == 1) addp = FF + (size_t)n*4*HW;
                       else if (l == 2) addp = ff02 + (size_t)n*2*HW; }
    else if (t == 1) { if (l == 0) addp = FB + (size_t)n*4*HW;
                       else if (l == 2) addp = FF + (size_t)n*4*HW + 2*HW; }
    else             { if (l == 0) addp = fb20 + (size_t)n*2*HW;
                       else if (l == 1) addp = FB + (size_t)n*4*HW + 2*HW; }
    float addx = addp ? addp[hw] : 0.f;
    float addy = addp ? addp[HW + hw] : 0.f;

    float refx = Ref[(((size_t)n * LQTOT + q) * NLVL + l) * 2 + 0];
    float refy = Ref[(((size_t)n * LQTOT + q) * NLVL + l) * 2 + 1];
    const float* vimg = Vconv + ((size_t)(n * TT + l) * CIN + head * DHEAD) * HW;

    for (int p = 0; p < NPTS; ++p) {
      const float* op = Oconv +
          ((size_t)g * (NHEAD*NLVL*NPTS*2) + ((head * NLVL + l) * NPTS + p) * 2) * HW + hw;
      float ox = op[0]  + addx;
      float oy = op[HW] + addy;
      float px = refx * WW + ox - 0.5f;   // (ref + off/W)*W - 0.5
      float py = refy * HH + oy - 0.5f;
      float aw = lg[l * NPTS + p] * inv;

      float x0f = floorf(px), y0f = floorf(py);
      int x0 = (int)x0f, y0 = (int)y0f;
      float wx = px - x0f, wy = py - y0f;
      const float cwt[4] = { (1-wx)*(1-wy), wx*(1-wy), (1-wx)*wy, wx*wy };
      const int cx[4] = { x0, x0+1, x0,   x0+1 };
      const int cy[4] = { y0, y0,   y0+1, y0+1 };
#pragma unroll
      for (int cc = 0; cc < 4; ++cc) {
        int xi = cx[cc], yi = cy[cc];
        if (xi >= 0 && xi < WW && yi >= 0 && yi < HH) {
          float wgt = aw * cwt[cc];
          const float* ip = vimg + yi * WW + xi;
#pragma unroll
          for (int d = 0; d < DHEAD; ++d) acc[d] += wgt * ip[d * HW];
        }
      }
    }
  }

  float* outp = Out + ((size_t)g * CIN + head * DHEAD) * HW + hw;
#pragma unroll
  for (int d = 0; d < DHEAD; ++d) outp[d * HW] = acc[d];
}

// -------------------------------------------------------------------------
extern "C" void kernel_launch(void* const* d_in, const int* in_sizes, int n_in,
                              void* d_out, int out_size, void* d_ws, size_t ws_size,
                              hipStream_t stream) {
  const float* query  = (const float*)d_in[0];
  const float* inputf = (const float*)d_in[1];
  const float* refpts = (const float*)d_in[2];
  const float* flow_f = (const float*)d_in[3];
  const float* flow_b = (const float*)d_in[4];
  const float* w_off  = (const float*)d_in[5];
  const float* b_off  = (const float*)d_in[6];
  const float* w_attn = (const float*)d_in[7];
  const float* b_attn = (const float*)d_in[8];
  const float* w_val  = (const float*)d_in[9];
  const float* b_val  = (const float*)d_in[10];
  const float* w_out  = (const float*)d_in[11];
  const float* b_out  = (const float*)d_in[12];
  float* out = (float*)d_out;

  float* ws = (float*)d_ws;
  float* conv_val  = ws;                                    // 6*128*2304
  float* conv_off  = conv_val  + (size_t)GG * 128 * HW;     // 6*192*2304
  float* conv_attn = conv_off  + (size_t)GG * 192 * HW;     // 6*96*2304
  float* ff02      = conv_attn + (size_t)GG *  96 * HW;     // 2*2*2304
  float* fb20      = ff02      + (size_t)NN * 2 * HW;       // 2*2*2304
  float* attn_out  = fb20      + (size_t)NN * 2 * HW;       // 6*128*2304

  dim3 blk(256);            // 8 waves/block, one 16-pixel M-tile each
  dim3 gv(108, 128 / 16);   // 13824 / (16*8) = 108 M-blocks
  dim3 go(108, 192 / 16);
  dim3 ga(108,  96 / 16);

  conv3x3_wmma<<<gv, blk, 0, stream>>>(inputf, w_val,  b_val,  conv_val,  128);
  conv3x3_wmma<<<go, blk, 0, stream>>>(query,  w_off,  b_off,  conv_off,  192);
  conv3x3_wmma<<<ga, blk, 0, stream>>>(query,  w_attn, b_attn, conv_attn,  96);

  flow_compose<<<(NN * HW + 255) / 256, 256, 0, stream>>>(flow_f, flow_b, ff02, fb20);

  deform_sample<<<(NN * LQTOT * NHEAD + 255) / 256, 256, 0, stream>>>(
      conv_val, conv_off, conv_attn, refpts, flow_f, flow_b, ff02, fb20, attn_out);

  conv3x3_wmma<<<gv, blk, 0, stream>>>(attn_out, w_out, b_out, out, 128);
}
